// InfiniAttention_65687229825607
// MI455X (gfx1250) — matrix operand order repaired and checked by reading, not compiled, not run on hardware
//
#include <hip/hip_runtime.h>

#define BB 2
#define HH 16
#define SS 2048
#define DD 64

typedef __attribute__((ext_vector_type(16))) __bf16 v16bf;
typedef __attribute__((ext_vector_type(8)))  __bf16 v8bf;
typedef __attribute__((ext_vector_type(8)))  float  v8f;

__device__ __forceinline__ v8f wmma_bf16(v16bf a, v16bf b, v8f c) {
  // (neg_a, A, neg_b, B, c_mod, C, reuse_a, reuse_b)
  return __builtin_amdgcn_wmma_f32_16x16x32_bf16(false, a, false, b, (short)0, c,
                                                 false, false);
}

__device__ __forceinline__ v8f vzero8() {
  v8f x;
#pragma unroll
  for (int i = 0; i < 8; ++i) x[i] = 0.f;
  return x;
}

__global__ __launch_bounds__(256) void infini_attn_kernel(
    const float* __restrict__ Q, const float* __restrict__ K,
    const float* __restrict__ V, const float* __restrict__ M,
    const float* __restrict__ Z, const float* __restrict__ Beta,
    float* __restrict__ Out) {
  // Double-buffered K/V staging + M/z + per-wave P transpose scratch
  __shared__ __align__(32) __bf16 Klds[2][32 * 64];   // row-major [key_local][d]
  __shared__ __align__(32) __bf16 Vlds[2][32 * 64];   // WMMA B-layout per 16-col tile
  __shared__ __align__(32) __bf16 Mlds[64 * 64];      // WMMA B-layout, (c,e) tiles
  __shared__ float Zlds[DD];
  __shared__ float Plds[8][16 * 32];

  const int tid  = threadIdx.x;
  const int wave = tid >> 5;
  const int lane = tid & 31;
  const int half = lane >> 4;
  const int colL = lane & 15;

  const int b = blockIdx.z, h = blockIdx.y;
  const int qbase = blockIdx.x * 128;
  const int qrow  = qbase + wave * 16;

  const size_t bh = (size_t)b * HH + h;
  const float* Qg = Q + bh * SS * DD;
  const float* Kg = K + bh * SS * DD;
  const float* Vg = V + bh * SS * DD;
  const float* Mg = M + bh * DD * DD;
  const float* Zg = Z + bh * DD;
  float*       Og = Out + bh * SS * DD;

  const float NEG_INF = -__builtin_inff();

  // ---- stage z and memory matrix M (fp32 -> bf16, pre-swizzled to B-layout) ----
  if (tid < DD) Zlds[tid] = Zg[tid];
  {
    int base = tid * 16;              // 64*64 elems / 256 threads = 16 each
    int d    = base >> 6;
    int c0m  = base & 63;
    const float4* mg4 = (const float4*)(Mg + base);
#pragma unroll
    for (int q4 = 0; q4 < 4; ++q4) {
      float4 mv = mg4[q4];
      float tmp[4] = {mv.x, mv.y, mv.z, mv.w};
#pragma unroll
      for (int u = 0; u < 4; ++u) {
        int colg = c0m + q4 * 4 + u;
        int cc = d >> 5, lh = (d >> 4) & 1, ii = d & 15;
        int e = colg >> 4, ln = lh * 16 + (colg & 15);
        Mlds[((cc * 4 + e) * 32 + ln) * 16 + ii] = (__bf16)tmp[u];
      }
    }
  }
  __syncthreads();

  // ---- per-wave: load 16x64 Q tile, build bf16 A-fragments for q and phi(q) ----
  const float QSCALE = 0.125f * 1.44269504088896340736f;  // 1/sqrt(D) * log2(e)
  v16bf qa[2], pha[2];
  float denom = 0.f;
  {
    const float* qr = Qg + (size_t)(qrow + colL) * DD;
#pragma unroll
    for (int f = 0; f < 2; ++f) {
#pragma unroll
      for (int c2 = 0; c2 < 2; ++c2) {
        int d0 = f * 32 + half * 16 + c2 * 8;
        float4 a0 = *(const float4*)(qr + d0);
        float4 a1 = *(const float4*)(qr + d0 + 4);
        float tmp[8] = {a0.x, a0.y, a0.z, a0.w, a1.x, a1.y, a1.z, a1.w};
#pragma unroll
        for (int j = 0; j < 8; ++j) {
          float x = tmp[j];
          qa[f][c2 * 8 + j] = (__bf16)(x * QSCALE);
          float ph = x > 0.f ? x + 1.f : __expf(x);   // elu(x)+1
          pha[f][c2 * 8 + j] = (__bf16)ph;
          denom += ph * Zlds[d0 + j];
        }
      }
    }
  }
  denom += __shfl_xor(denom, 16);   // combine the two 32-wide d halves per row

  // ---- compressive memory read: numer = phi(q) @ M  (8 WMMAs) ----
  v8f nm[4];
#pragma unroll
  for (int e = 0; e < 4; ++e) {
    v16bf mb0 = *(const v16bf*)&Mlds[((0 * 4 + e) * 32 + lane) * 16];
    v16bf mb1 = *(const v16bf*)&Mlds[((1 * 4 + e) * 32 + lane) * 16];
    nm[e] = wmma_bf16(pha[1], mb1, wmma_bf16(pha[0], mb0, vzero8()));
  }

  // constant all-ones B fragment: P @ 1 gives softmax row-sums in C-layout
  v16bf onesb;
#pragma unroll
  for (int i = 0; i < 16; ++i) onesb[i] = (__bf16)1.0f;

  // ---- flash-attention online-softmax loop over 32-key tiles ----
  float mrow[8];
  v8f lC = vzero8();
  v8f acc[4];
#pragma unroll
  for (int r = 0; r < 8; ++r) mrow[r] = NEG_INF;
#pragma unroll
  for (int e = 0; e < 4; ++e) acc[e] = vzero8();

  const int nT     = qbase / 32 + 4;   // tiles needed by this block's 128 rows
  const int myDiag = qrow >> 5;        // the single partially-masked tile

  // staging thread mapping (coalesced 8-float chunks)
  const int srow = tid >> 3;
  const int sc0  = (tid & 7) * 8;
  float kt[8], vt[8];

  auto load_regs = [&](int t) {
    const float* kp = Kg + (size_t)(t + srow) * DD + sc0;
    const float* vp = Vg + (size_t)(t + srow) * DD + sc0;
    float4 k0 = *(const float4*)(kp);
    float4 k1 = *(const float4*)(kp + 4);
    kt[0] = k0.x; kt[1] = k0.y; kt[2] = k0.z; kt[3] = k0.w;
    kt[4] = k1.x; kt[5] = k1.y; kt[6] = k1.z; kt[7] = k1.w;
    float4 v0 = *(const float4*)(vp);
    float4 v1 = *(const float4*)(vp + 4);
    vt[0] = v0.x; vt[1] = v0.y; vt[2] = v0.z; vt[3] = v0.w;
    vt[4] = v1.x; vt[5] = v1.y; vt[6] = v1.z; vt[7] = v1.w;
  };
  auto store_lds = [&](int pp) {
    v8bf kv;
#pragma unroll
    for (int j = 0; j < 8; ++j) kv[j] = (__bf16)kt[j];
    *(v8bf*)&Klds[pp][srow * 64 + sc0] = kv;       // one ds_store_b128
#pragma unroll
    for (int j = 0; j < 8; ++j) {
      int colg = sc0 + j;
      int e = colg >> 4, ln = (srow & 16) + (colg & 15), ii = srow & 15;
      Vlds[pp][(e * 32 + ln) * 16 + ii] = (__bf16)vt[j];
    }
  };

  // prologue: stage tile 0
  load_regs(0);
  store_lds(0);
  __syncthreads();

  int pbuf = 0;
  for (int it = 0; it < nT; ++it) {
    const bool more = (it + 1 < nT);
    if (more) load_regs((it + 1) * 32);   // prefetch next tile (overlaps WMMAs)

    if (it <= myDiag) {                   // uniform per wave
      const int t = it * 32;
      // scores: S = (q*scale) @ K^T for two 16-key blocks (4 WMMAs)
      const __bf16* Kb   = Klds[pbuf];
      const __bf16* kb0p = &Kb[colL * 64 + half * 16];
      const __bf16* kb1p = &Kb[(16 + colL) * 64 + half * 16];
      v16bf k00 = *(const v16bf*)(kb0p);
      v16bf k01 = *(const v16bf*)(kb0p + 32);
      v16bf k10 = *(const v16bf*)(kb1p);
      v16bf k11 = *(const v16bf*)(kb1p + 32);
      v8f s0 = wmma_bf16(qa[1], k01, wmma_bf16(qa[0], k00, vzero8()));
      v8f s1 = wmma_bf16(qa[1], k11, wmma_bf16(qa[0], k10, vzero8()));

      float p0[8], p1[8], bm[8];
      if (it == myDiag) {
        // diagonal tile: apply causal mask
        const int key0 = t + colL, key1 = t + 16 + colL;
#pragma unroll
        for (int r = 0; r < 8; ++r) {
          int rowg = qrow + r + 8 * half;
          float x0 = (key0 <= rowg) ? s0[r] : NEG_INF;
          float x1 = (key1 <= rowg) ? s1[r] : NEG_INF;
          p0[r] = x0; p1[r] = x1;
          bm[r] = fmaxf(x0, x1);
        }
      } else {
        // interior tile: fully unmasked fast path
#pragma unroll
        for (int r = 0; r < 8; ++r) {
          p0[r] = s0[r]; p1[r] = s1[r];
          bm[r] = fmaxf(s0[r], s1[r]);
        }
      }
#pragma unroll
      for (int off = 1; off < 16; off <<= 1)
#pragma unroll
        for (int r = 0; r < 8; ++r) bm[r] = fmaxf(bm[r], __shfl_xor(bm[r], off));

      float alpha[8];
#pragma unroll
      for (int r = 0; r < 8; ++r) {
        float mn = fmaxf(mrow[r], bm[r]);
        alpha[r] = exp2f(mrow[r] - mn);   // scores are in log2 units
        mrow[r]  = mn;
        p0[r] = exp2f(p0[r] - mn);
        p1[r] = exp2f(p1[r] - mn);
      }
#pragma unroll
      for (int r = 0; r < 8; ++r) lC[r] *= alpha[r];
#pragma unroll
      for (int e = 0; e < 4; ++e)
#pragma unroll
        for (int r = 0; r < 8; ++r) acc[e][r] *= alpha[r];

      // transpose P (C-layout -> A-layout) through per-wave LDS scratch
      float* Pw = Plds[wave];
#pragma unroll
      for (int r = 0; r < 8; ++r) {
        int rowL = r + 8 * half;
        Pw[rowL * 32 + colL]      = p0[r];
        Pw[rowL * 32 + 16 + colL] = p1[r];
      }
      v16bf pf;
#pragma unroll
      for (int p2 = 0; p2 < 8; ++p2) {
        int g = p2 >> 2, jj = (p2 & 3) * 2;
        int kk = half * 16 + g * 8 + jj;
        float2 pv = *(const float2*)&Pw[colL * 32 + kk];
        pf[2 * p2]     = (__bf16)pv.x;
        pf[2 * p2 + 1] = (__bf16)pv.y;
      }
      // running denominator: l += P @ 1 (one WMMA, no cross-lane shuffles)
      lC = wmma_bf16(pf, onesb, lC);
      // O += P @ V  (4 WMMAs)
      const __bf16* Vb = Vlds[pbuf];
#pragma unroll
      for (int e = 0; e < 4; ++e) {
        v16bf vb = *(const v16bf*)&Vb[(e * 32 + lane) * 16];
        acc[e] = wmma_bf16(pf, vb, acc[e]);
      }
    }

    if (more) store_lds(pbuf ^ 1);        // stage next tile into the other buffer
    __syncthreads();                      // single barrier per iteration
    pbuf ^= 1;
  }

  // ---- gated combine + store ----
  float gate = 1.f / (1.f + __expf(-Beta[h]));
  float dC[8];
#pragma unroll
  for (int r = 0; r < 8; ++r) dC[r] = __shfl(denom, r + 8 * half);
#pragma unroll
  for (int r = 0; r < 8; ++r) {
    float li = 1.f / lC[r];
    float md = 1.f / (dC[r] + 1e-6f);
    int rowg = qrow + r + 8 * half;
    float* op = Og + (size_t)rowg * DD + colL;
#pragma unroll
    for (int e = 0; e < 4; ++e)
      op[e * 16] = gate * acc[e][r] * li + (1.f - gate) * nm[e][r] * md;
  }
}

extern "C" void kernel_launch(void* const* d_in, const int* in_sizes, int n_in,
                              void* d_out, int out_size, void* d_ws, size_t ws_size,
                              hipStream_t stream) {
  (void)in_sizes; (void)n_in; (void)out_size; (void)d_ws; (void)ws_size;
  const float* q    = (const float*)d_in[0];
  const float* k    = (const float*)d_in[1];
  const float* v    = (const float*)d_in[2];
  const float* M    = (const float*)d_in[3];
  const float* z    = (const float*)d_in[4];
  const float* beta = (const float*)d_in[5];
  float* out = (float*)d_out;
  dim3 grid(SS / 128, HH, BB);
  dim3 block(256);
  infini_attn_kernel<<<grid, block, 0, stream>>>(q, k, v, M, z, beta, out);
}
